// LSTM_Layer_65223373357304
// MI455X (gfx1250) — matrix operand order repaired and checked by reading, not compiled, not run on hardware
//
#include <hip/hip_runtime.h>

// ---------------------------------------------------------------------------
// LSTM layer for MI455X (gfx1250), persistent-kernel formulation.
//   B=128, T=512, D=512, U=512
//   z = x_t @ Wx + h @ Wh + b   (K = D+U = 1024, fused)
//   WMMA: v_wmma_f32_16x16x32_bf16, f32 accumulate.
//
// Phase 0 (parallel): convert x -> bf16 in workspace (if ws large enough).
// Phase 1 (persistent): 32 blocks (one per 16-wide u-tile), 8 wave32 each.
//   Wave m computes rows 16m..16m+15 for all 4 gates of its u-tile.
//   Weights LDS-resident (128 KB of 320 KB/WGP), pre-swizzled to the
//   B-fragment lane layout. h double-buffered in bf16; grid-wide sense
//   barrier per timestep (recurrence is latency-bound, not BW/FLOP-bound:
//   275 GFLOP total, 264 MB unique bytes ~ 11 us at 23.3 TB/s).
// ---------------------------------------------------------------------------

typedef __attribute__((ext_vector_type(16))) __bf16 v16bf;
typedef __attribute__((ext_vector_type(8)))  __bf16 v8bf;
typedef __attribute__((ext_vector_type(4)))  __bf16 v4bf;
typedef __attribute__((ext_vector_type(8)))  float  v8f;
typedef __attribute__((ext_vector_type(4)))  float  f32x4;

#define BB 128
#define TT 512
#define DD 512
#define UU 512
#define GATES 4
#define KTOT (DD + UU)          // 1024
#define NCHUNK (KTOT / 32)      // 32 k-chunks of 32
#define NBLOCKS (UU / 16)       // 32 u-tiles

// ---- grid-wide barrier (agent-scope atomics, sense = generation) ----------
__device__ __forceinline__ void grid_barrier(unsigned* cnt, unsigned* gen,
                                             unsigned nblocks) {
    __syncthreads();
    __threadfence();   // make H/C stores device-visible before arrive
    if (threadIdx.x == 0) {
        unsigned g = __hip_atomic_load(gen, __ATOMIC_ACQUIRE,
                                       __HIP_MEMORY_SCOPE_AGENT);
        unsigned prev = __hip_atomic_fetch_add(cnt, 1u, __ATOMIC_ACQ_REL,
                                               __HIP_MEMORY_SCOPE_AGENT);
        if (prev == nblocks - 1u) {
            __hip_atomic_store(cnt, 0u, __ATOMIC_RELAXED,
                               __HIP_MEMORY_SCOPE_AGENT);
            __hip_atomic_store(gen, g + 1u, __ATOMIC_RELEASE,
                               __HIP_MEMORY_SCOPE_AGENT);
        } else {
            while (__hip_atomic_load(gen, __ATOMIC_ACQUIRE,
                                     __HIP_MEMORY_SCOPE_AGENT) == g) {
                __builtin_amdgcn_s_sleep(2);
            }
        }
    }
    __syncthreads();
}

__device__ __forceinline__ float fast_sigmoid(float v) {
    return 1.0f / (1.0f + __expf(-v));
}

// ---- phase 0: x (f32, 128 MB) -> bf16 copy (64 MB), fully parallel --------
__global__ void __launch_bounds__(256)
convert_x_bf16(const float* __restrict__ x, __bf16* __restrict__ xb,
               size_t n4) {
    size_t i = (size_t)blockIdx.x * blockDim.x + threadIdx.x;
    const size_t stride = (size_t)gridDim.x * blockDim.x;
    for (; i < n4; i += stride) {
        const f32x4 f = *(const f32x4*)(x + i * 4);
        v4bf o;
#pragma unroll
        for (int k = 0; k < 4; ++k) o[k] = (__bf16)f[k];
        *(v4bf*)(xb + i * 4) = o;
    }
}

// B-fragment LDS layout: [gate][chunk][lane][16 bf16], 32B/lane contiguous.
// ISA B layout (32x16, bf16): lanes 0-15 hold K 0..7,16..23, lanes 16-31 hold
// K 8..15,24..31 of the chunk, column N = lane%16.
template <bool XBF>
__global__ void __launch_bounds__(256)
lstm_persistent(const float* __restrict__ x,        // [B][T][D] f32
                const __bf16* __restrict__ xb,      // [B][T][D] bf16 (opt)
                const float* __restrict__ h0,
                const float* __restrict__ c0,
                const float* __restrict__ Wx,
                const float* __restrict__ Wh,
                const float* __restrict__ bvec,
                float* __restrict__ out,            // [T][B][U]
                float* __restrict__ Cst,            // [B][U] f32
                __bf16* __restrict__ H0buf,         // [B][U] bf16
                __bf16* __restrict__ H1buf,         // [B][U] bf16
                unsigned* cnt, unsigned* gen) {
    extern __shared__ __bf16 Bsm[];                 // [4][32][32][16] = 128 KB

    const int lane   = threadIdx.x & 31;
    const int wave   = threadIdx.x >> 5;
    const int laneHi = (lane >> 4) & 1;
    const int lmod   = lane & 15;
    const int row0   = wave * 16;                   // batch-row tile
    const int col0   = blockIdx.x * 16;             // u-tile

    // ---- stage weights into LDS, pre-swizzled to B-fragment layout -------
    for (int s = threadIdx.x; s < GATES * NCHUNK * 32; s += 256) {
        const int g = s >> 10;                      // /(32*32)
        const int c = (s >> 5) & 31;
        const int l = s & 31;
        const int N = l & 15;
        const int kofs = (l >= 16) ? 8 : 0;
        const int gcol = g * UU + col0 + N;
        __bf16* dst = &Bsm[(size_t)s * 16];
#pragma unroll
        for (int e = 0; e < 16; ++e) {
            const int K = c * 32 + kofs + e + ((e >= 8) ? 8 : 0);
            const float v = (K < DD)
                ? Wx[(size_t)K * (4 * UU) + gcol]
                : Wh[(size_t)(K - DD) * (4 * UU) + gcol];
            dst[e] = (__bf16)v;
        }
    }

    // ---- init block-owned columns of C and H[0] ---------------------------
    for (int idx = threadIdx.x; idx < BB * 16; idx += 256) {
        const int r = idx >> 4;
        const int n = idx & 15;
        const size_t ci = (size_t)r * UU + col0 + n;
        Cst[ci]   = c0[ci];
        H0buf[ci] = (__bf16)h0[ci];
    }

    // per-lane gate bias (same for all 8 accumulator rows of a lane)
    float bias[GATES];
#pragma unroll
    for (int g = 0; g < GATES; ++g) bias[g] = bvec[g * UU + col0 + lmod];

    grid_barrier(cnt, gen, NBLOCKS);

    const float*  xrow_base  = x  + (size_t)(row0 + lmod) * (TT * DD);
    const __bf16* xbrow_base = xb + (size_t)(row0 + lmod) * (TT * DD);

    for (int t = 0; t < TT; ++t) {
        const __bf16* Hc = (t & 1) ? H1buf : H0buf;
        __bf16*       Hn = (t & 1) ? H0buf : H1buf;

        v8f acc[GATES];
#pragma unroll
        for (int g = 0; g < GATES; ++g)
#pragma unroll
            for (int e = 0; e < 8; ++e) acc[g][e] = bias[g];

        // ---- K = 0..511 : x_t @ Wx ---------------------------------------
        const float*  xrow  = xrow_base  + (size_t)t * DD;
        const __bf16* xbrow = xbrow_base + (size_t)t * DD;
        for (int c = 0; c < NCHUNK / 2; ++c) {
            const int kb = c * 32 + laneHi * 8;
            v16bf a;
            if (XBF) {
                // bf16 path: 2x b128 loads, pure register packing
                __builtin_prefetch(xbrow + kb + 128, 0, 1);
                const v8bf lo = *(const v8bf*)(xbrow + kb);
                const v8bf hi = *(const v8bf*)(xbrow + kb + 16);
#pragma unroll
                for (int i = 0; i < 8; ++i) { a[i] = lo[i]; a[i + 8] = hi[i]; }
            } else {
                // f32 fallback: 4x b128 loads + cvt chain
                __builtin_prefetch(xrow + kb + 256, 0, 1);
                const f32x4 f0 = *(const f32x4*)(xrow + kb);
                const f32x4 f1 = *(const f32x4*)(xrow + kb + 4);
                const f32x4 f2 = *(const f32x4*)(xrow + kb + 16);
                const f32x4 f3 = *(const f32x4*)(xrow + kb + 20);
#pragma unroll
                for (int i = 0; i < 4; ++i) {
                    a[i]      = (__bf16)f0[i];
                    a[i + 4]  = (__bf16)f1[i];
                    a[i + 8]  = (__bf16)f2[i];
                    a[i + 12] = (__bf16)f3[i];
                }
            }
#pragma unroll
            for (int g = 0; g < GATES; ++g) {
                const __bf16* bp =
                    &Bsm[((((size_t)g * NCHUNK + c) * 32) + lane) * 16];
                const v8bf b0 = *(const v8bf*)bp;
                const v8bf b1 = *(const v8bf*)(bp + 8);
                v16bf bb;
#pragma unroll
                for (int i = 0; i < 8; ++i) { bb[i] = b0[i]; bb[i + 8] = b1[i]; }
                acc[g] = __builtin_amdgcn_wmma_f32_16x16x32_bf16(
                    false, a, false, bb, (short)0, acc[g], false, false);
            }
        }

        // ---- K = 512..1023 : h @ Wh --------------------------------------
        const __bf16* hrow = Hc + (size_t)(row0 + lmod) * UU;
        for (int c = 0; c < NCHUNK / 2; ++c) {
            const int kb = c * 32 + laneHi * 8;
            const v8bf lo = *(const v8bf*)(hrow + kb);
            const v8bf hi = *(const v8bf*)(hrow + kb + 16);
            v16bf a;
#pragma unroll
            for (int i = 0; i < 8; ++i) { a[i] = lo[i]; a[i + 8] = hi[i]; }
#pragma unroll
            for (int g = 0; g < GATES; ++g) {
                const int cc = NCHUNK / 2 + c;    // absolute chunk 16..31
                const __bf16* bp =
                    &Bsm[((((size_t)g * NCHUNK + cc) * 32) + lane) * 16];
                const v8bf b0 = *(const v8bf*)bp;
                const v8bf b1 = *(const v8bf*)(bp + 8);
                v16bf bb;
#pragma unroll
                for (int i = 0; i < 8; ++i) { bb[i] = b0[i]; bb[i + 8] = b1[i]; }
                acc[g] = __builtin_amdgcn_wmma_f32_16x16x32_bf16(
                    false, a, false, bb, (short)0, acc[g], false, false);
            }
        }

        // ---- pointwise gate math directly on WMMA C-layout ----------------
        // element e of acc: M = e + 8*laneHi, N = lmod
#pragma unroll
        for (int e = 0; e < 8; ++e) {
            const int M = e + laneHi * 8;
            const size_t ci = (size_t)(row0 + M) * UU + col0 + lmod;
            const float cold = Cst[ci];
            const float iv = fast_sigmoid(acc[0][e]);
            const float fv = fast_sigmoid(acc[1][e]);
            const float gv = tanhf(acc[2][e]);
            const float ov = fast_sigmoid(acc[3][e]);
            const float cn = fv * cold + iv * gv;
            const float hn = ov * tanhf(cn);
            Cst[ci] = cn;
            Hn[ci]  = (__bf16)hn;
            out[(size_t)t * (BB * UU) + ci] = hn;
        }

        grid_barrier(cnt, gen, NBLOCKS);
    }
}

// ---------------------------------------------------------------------------
extern "C" void kernel_launch(void* const* d_in, const int* in_sizes, int n_in,
                              void* d_out, int out_size, void* d_ws,
                              size_t ws_size, hipStream_t stream) {
    (void)in_sizes; (void)n_in; (void)out_size;
    const float* x  = (const float*)d_in[0];
    const float* h0 = (const float*)d_in[1];
    const float* c0 = (const float*)d_in[2];
    const float* Wx = (const float*)d_in[3];
    const float* Wh = (const float*)d_in[4];
    const float* bv = (const float*)d_in[5];
    float* out = (float*)d_out;

    unsigned char* ws = (unsigned char*)d_ws;
    unsigned* cnt = (unsigned*)(ws + 0);
    unsigned* gen = (unsigned*)(ws + 64);

    size_t off = 256;
    float* Cst = (float*)(ws + off);        off += (size_t)BB * UU * 4; // 256 KB
    __bf16* H0b = (__bf16*)(ws + off);      off += (size_t)BB * UU * 2; // 128 KB
    __bf16* H1b = (__bf16*)(ws + off);      off += (size_t)BB * UU * 2; // 128 KB
    off = (off + 255) & ~(size_t)255;
    __bf16* Xbf = (__bf16*)(ws + off);
    const size_t need_xbf = off + (size_t)BB * TT * DD * 2;             // +64 MB
    const bool use_xbf = (ws_size >= need_xbf);

    // workspace is poisoned 0xAA by the harness; barrier words must be 0
    hipMemsetAsync(ws, 0, 256, stream);

    const size_t smem =
        (size_t)GATES * NCHUNK * 32 * 16 * sizeof(__bf16);              // 128 KB

    if (use_xbf) {
        const size_t n4 = (size_t)BB * TT * DD / 4;                     // 8 M
        convert_x_bf16<<<4096, 256, 0, stream>>>(x, Xbf, n4);
        lstm_persistent<true><<<NBLOCKS, 256, smem, stream>>>(
            x, Xbf, h0, c0, Wx, Wh, bv, out, Cst, H0b, H1b, cnt, gen);
    } else {
        lstm_persistent<false><<<NBLOCKS, 256, smem, stream>>>(
            x, (const __bf16*)nullptr, h0, c0, Wx, Wh, bv, out,
            Cst, H0b, H1b, cnt, gen);
    }
}
